// MoEOpModel_41540923687450
// MI455X (gfx1250) — matrix operand order repaired and checked by reading, not compiled, not run on hardware
//
#include <hip/hip_runtime.h>
#include <math.h>

// Problem dims (fixed by the reference)
constexpr int kT = 4096;   // tokens
constexpr int kH = 1024;   // hidden
constexpr int kI = 512;    // intermediate
constexpr int kE = 64;     // experts
// TOP_K = 2

// ---- vector types ----
typedef __attribute__((ext_vector_type(16))) __bf16          bf16x16;
typedef __attribute__((ext_vector_type(8)))  __bf16          bf16x8;
typedef __attribute__((ext_vector_type(8)))  float           f32x8;
typedef __attribute__((ext_vector_type(4)))  float           f32x4;
typedef __attribute__((ext_vector_type(8)))  unsigned short  u16x8;
typedef __attribute__((ext_vector_type(8)))  unsigned int    u32x8;
typedef __attribute__((ext_vector_type(2)))  unsigned int    u32x2;

// Packed fp32x2 -> bf16x2 using the CDNA5 VALU packed converter (1 op / 2 elems)
// D[15:0] = bf16(a), D[31:16] = bf16(b)
static __device__ inline unsigned cvt_pk_bf16(float a, float b) {
    unsigned r;
    asm("v_cvt_pk_bf16_f32 %0, %1, %2" : "=v"(r) : "v"(a), "v"(b));
    return r;
}

// Scalar fp32 -> bf16 bits (RNE) for the few per-tile activation stores
static __device__ inline unsigned short f2bf_bits(float f) {
    unsigned u = __builtin_bit_cast(unsigned, f);
    u += 0x7FFFu + ((u >> 16) & 1u);
    return (unsigned short)(u >> 16);
}

// Load 8 contiguous bf16 (stored as u16) from LDS, 16B aligned
static __device__ inline bf16x8 lds_ld8(const unsigned short* p) {
    return __builtin_bit_cast(bf16x8, *(const u16x8*)p);
}

// A-fragment (16x32 bf16): lane's row pointer already offset by k;
// 16 contiguous K values (two 8-elem chunks at +0 and +8).
static __device__ inline bf16x16 load_a_frag(const unsigned short* p) {
    bf16x8 lo = lds_ld8(p);
    bf16x8 hi = lds_ld8(p + 8);
    return __builtin_shufflevector(lo, hi, 0,1,2,3,4,5,6,7,8,9,10,11,12,13,14,15);
}

// B-fragment (32x16 bf16) from a row-major fp32 matrix treated as B^T:
// lane holds 16 contiguous K values of column (lane&15), K base = (lane>>4)*16.
// Packed conversion: 8 VALU ops per fragment.
static __device__ inline bf16x16 load_b_f32(const float* __restrict__ p) {
    u32x8 r;
#pragma unroll
    for (int c = 0; c < 4; ++c) {
        f32x4 f = *(const f32x4*)(p + 4 * c);
        r[2*c+0] = cvt_pk_bf16(f[0], f[1]);
        r[2*c+1] = cvt_pk_bf16(f[2], f[3]);
    }
    return __builtin_bit_cast(bf16x16, r);
}

// ---------------------------------------------------------------------------
// Kernel 0: zero output + expert counters
// ---------------------------------------------------------------------------
__global__ void moe_zero(float* __restrict__ out, int* __restrict__ cnt, int n) {
    int i = blockIdx.x * blockDim.x + threadIdx.x;
    if (i < n) out[i] = 0.0f;
    if (i < kE) cnt[i] = 0;
}

// ---------------------------------------------------------------------------
// Kernel 1: router — logits, top-2, renormalized weights, expert compaction
// One 64-thread block per token; thread e owns expert e's logit.
// ---------------------------------------------------------------------------
__global__ __launch_bounds__(64) void moe_router(
    const float* __restrict__ x, const float* __restrict__ gw,
    const float* __restrict__ gb,
    int* __restrict__ cnt, int* __restrict__ tok_list, float* __restrict__ wt_list)
{
    __shared__ float xs[kH];
    __shared__ float lg[kE];
    const int t = blockIdx.x;
    const int e = threadIdx.x;

    // stage x row in LDS (64 threads * 4 float4 each = 1024 floats)
    {
        const f32x4* src = (const f32x4*)(x + (size_t)t * kH);
        f32x4* dst = (f32x4*)xs;
#pragma unroll
        for (int i = 0; i < 4; ++i) dst[e + 64 * i] = src[e + 64 * i];
    }
    __syncthreads();

    const float* g = gw + (size_t)e * kH;
    float acc = 0.0f;
    for (int h = 0; h < kH; h += 4) {
        f32x4 gv = *(const f32x4*)(g + h);
        f32x4 xv = *(const f32x4*)(xs + h);
        acc += gv[0]*xv[0] + gv[1]*xv[1] + gv[2]*xv[2] + gv[3]*xv[3];
    }
    lg[e] = acc + gb[e];
    __syncthreads();

    if (e == 0) {
        // top-2 of logits == top-2 of softmax probs; renormalized weights need
        // only the two logits: w1 = 1/(1+exp(l2-l1)). Ties break to lower index.
        int i1 = 0; float l1 = lg[0];
        for (int i = 1; i < kE; ++i) if (lg[i] > l1) { l1 = lg[i]; i1 = i; }
        int i2 = (i1 == 0) ? 1 : 0; float l2 = lg[i2];
        for (int i = 0; i < kE; ++i)
            if (i != i1 && lg[i] > l2) { l2 = lg[i]; i2 = i; }
        float w1 = 1.0f / (1.0f + __expf(l2 - l1));
        float w2 = 1.0f - w1;

        int p1 = atomicAdd(&cnt[i1], 1);
        tok_list[i1 * kT + p1] = t;  wt_list[i1 * kT + p1] = w1;
        int p2 = atomicAdd(&cnt[i2], 1);
        tok_list[i2 * kT + p2] = t;  wt_list[i2 * kT + p2] = w2;
    }
}

// ---------------------------------------------------------------------------
// Kernel 2: per-expert SwiGLU FFN on a 16-token tile, bf16 WMMA, fp32 accum.
// grid = (E, T/16); blocks past the expert's token count exit immediately.
// ---------------------------------------------------------------------------
constexpr int kXStride = 1040;  // 1024 + 16 pad (bf16 elems) -> breaks bank alias
constexpr int kAStride = 528;   // 512 + 16 pad

__global__ __launch_bounds__(256) void moe_ffn(
    const float* __restrict__ x,
    const float* __restrict__ W1, const float* __restrict__ W2,
    const float* __restrict__ W3,
    const int* __restrict__ cnt, const int* __restrict__ tok_list,
    const float* __restrict__ wt_list,
    float* __restrict__ out)
{
    const int e    = blockIdx.x;
    const int n_e  = cnt[e];
    const int row0 = blockIdx.y * 16;
    if (row0 >= n_e) return;

    __shared__ __align__(16) unsigned short Xs[16 * kXStride]; // x tile, bf16
    __shared__ __align__(16) unsigned short As[16 * kAStride]; // act tile, bf16
    __shared__ int   s_tok[16];
    __shared__ float s_wt[16];

    const int tid  = threadIdx.x;
    const int lane = tid & 31;
    const int wave = tid >> 5;

    if (tid < 16) {
        int idx = row0 + tid;
        s_tok[tid] = (idx < n_e) ? tok_list[e * kT + idx] : -1;
        s_wt[tid]  = (idx < n_e) ? wt_list[e * kT + idx] : 0.0f;
    }
    __syncthreads();

    // ---- stage gathered x rows into LDS as bf16 (zero-fill padding rows) ----
    {
        const int row = tid >> 4;       // 0..15
        const int cg  = tid & 15;       // 16 column groups of 64
        const int t   = s_tok[row];
        unsigned short* dst = Xs + row * kXStride;
        if (t >= 0) {
            const float* src = x + (size_t)t * kH;
#pragma unroll 4
            for (int i = 0; i < 16; ++i) {
                int col = cg * 64 + i * 4;
                f32x4 v = *(const f32x4*)(src + col);
                u32x2 pk;
                pk[0] = cvt_pk_bf16(v[0], v[1]);
                pk[1] = cvt_pk_bf16(v[2], v[3]);
                *(u32x2*)(dst + col) = pk;       // 8B packed LDS store
            }
        } else {
#pragma unroll 4
            for (int i = 0; i < 16; ++i) {
                int col = cg * 64 + i * 4;
                *(u32x2*)(dst + col) = (u32x2){0u, 0u};
            }
        }
    }
    __syncthreads();

    const int r  = lane & 15;           // fragment row / B column
    const int ak = (lane >> 4) * 16;    // A-frag K base (contiguous, matches B)
    const int bk = (lane >> 4) * 16;    // B-frag K base (contiguous, matches A)
    const int mh = (lane >> 4) * 8;     // C/D: lanes 16..31 hold rows +8

    // ---- GEMM1 & GEMM3 + SiLU + route-weight -> activation tile in LDS ----
    {
        const float* w1e = W1 + (size_t)e * kI * kH;
        const float* w3e = W3 + (size_t)e * kI * kH;
        const unsigned short* arow = Xs + r * kXStride + ak;
#pragma unroll
        for (int j = 0; j < 4; ++j) {
            const int nbase = (wave + 8 * j) * 16;      // 32 N-tiles over I=512
            const float* b1row = w1e + (size_t)(nbase + r) * kH + bk;
            const float* b3row = w3e + (size_t)(nbase + r) * kH + bk;
            f32x8 acc1 = {}; f32x8 acc3 = {};
            for (int k = 0; k < kH; k += 32) {
                bf16x16 a  = load_a_frag(arow + k);
                bf16x16 b1 = load_b_f32(b1row + k);
                bf16x16 b3 = load_b_f32(b3row + k);
                acc1 = __builtin_amdgcn_wmma_f32_16x16x32_bf16(
                           false, a, false, b1, (short)0, acc1, false, false);
                acc3 = __builtin_amdgcn_wmma_f32_16x16x32_bf16(
                           false, a, false, b3, (short)0, acc3, false, false);
            }
            const int nc = nbase + r;   // column within I
#pragma unroll
            for (int i = 0; i < 8; ++i) {
                const int m = i + mh;   // token row within tile (C/D layout)
                float h1 = acc1[i];
                // silu(h) = h * sigmoid(h); fast v_rcp instead of IEEE divide
                float sil = h1 * __builtin_amdgcn_rcpf(1.0f + __expf(-h1));
                float av = sil * acc3[i] * s_wt[m];
                As[m * kAStride + nc] = f2bf_bits(av);
            }
        }
    }
    __syncthreads();

    // ---- GEMM2: out[t, :] += act @ W2[e]^T  (atomic combine over top-2) ----
    {
        const float* w2e = W2 + (size_t)e * kH * kI;
        const unsigned short* arow = As + r * kAStride + ak;
#pragma unroll
        for (int j = 0; j < 8; ++j) {
            const int nbase = (wave + 8 * j) * 16;      // 64 N-tiles over H=1024
            const float* brow = w2e + (size_t)(nbase + r) * kI + bk;
            f32x8 acc = {};
            for (int k = 0; k < kI; k += 32) {
                bf16x16 a = load_a_frag(arow + k);
                bf16x16 b = load_b_f32(brow + k);
                acc = __builtin_amdgcn_wmma_f32_16x16x32_bf16(
                          false, a, false, b, (short)0, acc, false, false);
            }
            const int n = nbase + r;
#pragma unroll
            for (int i = 0; i < 8; ++i) {
                const int m = i + mh;
                const int t = s_tok[m];
                if (t >= 0) atomicAdd(out + (size_t)t * kH + n, acc[i]);
            }
        }
    }
}

// ---------------------------------------------------------------------------
extern "C" void kernel_launch(void* const* d_in, const int* in_sizes, int n_in,
                              void* d_out, int out_size, void* d_ws, size_t ws_size,
                              hipStream_t stream)
{
    const float* x  = (const float*)d_in[0];
    const float* gw = (const float*)d_in[1];
    const float* gb = (const float*)d_in[2];
    const float* W1 = (const float*)d_in[3];
    const float* W2 = (const float*)d_in[4];
    const float* W3 = (const float*)d_in[5];
    float* out = (float*)d_out;

    // workspace: [cnt: kE ints][tok_list: kE*kT ints][wt_list: kE*kT floats]
    int*   cnt = (int*)d_ws;
    int*   tok = cnt + kE;
    float* wt  = (float*)(tok + (size_t)kE * kT);

    const int n = kT * kH;
    moe_zero<<<(n + 255) / 256, 256, 0, stream>>>(out, cnt, n);
    moe_router<<<kT, 64, 0, stream>>>(x, gw, gb, cnt, tok, wt);
    dim3 grid(kE, kT / 16);
    moe_ffn<<<grid, 256, 0, stream>>>(x, W1, W2, W3, cnt, tok, wt, out);
}
